// EventLinearEncoder_70944269795734
// MI455X (gfx1250) — matrix operand order repaired and checked by reading, not compiled, not run on hardware
//
#include <hip/hip_runtime.h>

// ---------------- problem constants (match reference) ----------------
#define NS   3
#define NE   8
#define GG   72        // NS*NS*NE triplet groups
#define EE   131072    // edges
#define DIN  256
#define DOUT 256
#define CAPc 2560      // per-group capacity
#define TM   64        // rows (edges) per block tile

typedef _Float16 v16h __attribute__((ext_vector_type(16)));
typedef float    v8f  __attribute__((ext_vector_type(8)));

// CDNA5 async global->LDS path (ASYNCcnt-tracked), guarded for toolchain support
#if defined(__has_builtin)
#if __has_builtin(__builtin_amdgcn_global_load_async_to_lds_b128) && \
    __has_builtin(__builtin_amdgcn_s_wait_asynccnt)
#define USE_ASYNC_LDS 1
#endif
#endif
#ifndef USE_ASYNC_LDS
#define USE_ASYNC_LDS 0
#endif

#if USE_ASYNC_LDS
// builtin wants: ( int __vector(4) __device__(AS1) *, int __vector(4) AS3 *, imm, imm )
typedef int v4i_vs __attribute__((vector_size(16)));
typedef __attribute__((address_space(1))) v4i_vs* gas_v4i;
typedef __attribute__((address_space(3))) v4i_vs* las_v4i;
#endif

// ---------------- phase 0: zero group counters ----------------
__global__ void zero_counts_kernel(int* __restrict__ counts) {
    if (threadIdx.x < GG) counts[threadIdx.x] = 0;
}

// ---------------- phase 1: routing + dispatch ----------------
__device__ __forceinline__ int argmax_n(const float* __restrict__ p, int n) {
    int bi = 0; float bv = p[0];
    #pragma unroll
    for (int i = 1; i < 8; ++i) {
        if (i < n) { float v = p[i]; if (v > bv) { bv = v; bi = i; } }
    }
    return bi;                               // first-max on ties, like jnp.argmax
}

__global__ void route_kernel(const float* __restrict__ node_type,   // [NN,NS]
                             const float* __restrict__ edge_types,  // [E,NE]
                             const int*   __restrict__ edge_index,  // [2,E]
                             int* __restrict__ counts,              // [G]
                             int* __restrict__ buckets)             // [G,CAP]
{
    int e = blockIdx.x * blockDim.x + threadIdx.x;
    if (e >= EE) return;
    int s = edge_index[e];
    int d = edge_index[EE + e];
    int st = argmax_n(node_type + (size_t)s * NS, NS);
    int dt = argmax_n(node_type + (size_t)d * NS, NS);
    int et = argmax_n(edge_types + (size_t)e * NE, NE);
    int g  = (st * NS + dt) * NE + et;
    int pos = atomicAdd(&counts[g], 1);
    if (pos < CAPc) buckets[g * CAPc + pos] = e;   // overflow prob ~0 (ref drops too)
}

// ---------------- helpers ----------------
__device__ __forceinline__ unsigned pk2h(float a, float b) {
    union { _Float16 h[2]; unsigned u; } cv;
    cv.h[0] = (_Float16)a; cv.h[1] = (_Float16)b;
    return cv.u;
}

// ---------------- phase 1b: one-time W fp32 -> fp16, pre-swizzled ----------------
// dst layout (halfs): Wh[g][kc(8)][ct(16)][lane(32)][slot(16)]
//   B-fragment order used here (contiguous k, same as A): lanes 0-15 hold K=0..15, lanes 16-31 hold
//   K=16..31; VGPR v holds K=2v,2v+1  ->  lane = lhk*16 + (n&15), slot = kl&15.
__global__ __launch_bounds__(256)
void wconvert_kernel(const float* __restrict__ W,          // [G,DIN,DOUT]
                     unsigned short* __restrict__ Wh)      // [G*DIN*DOUT] halfs
{
    // one (g, k2, n0..n0+3) quad per thread; k2 = k/2
    int flat4 = blockIdx.x * blockDim.x + threadIdx.x;     // 589824 total
    if (flat4 >= GG * (DIN / 2) * (DOUT / 4)) return;
    int g   = flat4 / ((DIN / 2) * (DOUT / 4));
    int rem = flat4 % ((DIN / 2) * (DOUT / 4));
    int k2  = rem / (DOUT / 4);
    int n0  = (rem % (DOUT / 4)) * 4;
    int k   = 2 * k2;

    const float* src0 = W + ((size_t)g * DIN + k)     * DOUT + n0;
    const float* src1 = W + ((size_t)g * DIN + k + 1) * DOUT + n0;
    float4 r0 = *(const float4*)src0;
    float4 r1 = *(const float4*)src1;
    float a0[4] = {r0.x, r0.y, r0.z, r0.w};
    float a1[4] = {r1.x, r1.y, r1.z, r1.w};

    int kc  = k >> 5;
    int kl  = k & 31;
    int lhk = kl >> 4;
    int s   = kl & 15;                                     // even
    unsigned* dst32 = (unsigned*)Wh;
    #pragma unroll
    for (int j = 0; j < 4; ++j) {
        int n    = n0 + j;
        int lane = lhk * 16 + (n & 15);
        size_t halfidx = ((((size_t)g * 8 + kc) * 16 + (n >> 4)) * 32 + lane) * 16 + s;
        dst32[halfidx >> 1] = pk2h(a0[j], a1[j]);          // (k, k+1) packed
    }
}

// ---------------- phase 2: grouped GEMM via WMMA ----------------
__global__ __launch_bounds__(256)
void grouped_gemm_kernel(const float* __restrict__ x,              // [E,DIN] fp32
                         const unsigned short* __restrict__ Wh,    // pre-swizzled fp16
                         const float* __restrict__ bias,           // [G,DOUT]
                         const int*   __restrict__ counts,         // [G]
                         const int*   __restrict__ buckets,        // [G,CAP]
                         float* __restrict__ out)                  // [E,DOUT]
{
    const int g = blockIdx.y;
    int cnt = counts[g]; if (cnt > CAPc) cnt = CAPc;
    const int row0 = blockIdx.x * TM;
    if (row0 >= cnt) return;                 // block-uniform early exit
    const int rows = (cnt - row0 < TM) ? (cnt - row0) : TM;

    __shared__ int eid[TM];
    __shared__ __align__(32) _Float16 As[4][32][16];    //  4 KB: 4 row-halves, frag layout
    __shared__ __align__(32) _Float16 Bs[16][32][16];   // 16 KB: 16 col-tiles, frag layout

    const int t    = threadIdx.x;
    const int lane = t & 31;
    const int wv   = t >> 5;        // wave id 0..7 -> columns [32*wv, 32*wv+32)
    const int lh   = lane >> 4;     // lane half
    const int ln   = lane & 15;

    if (t < TM) eid[t] = (t < rows) ? buckets[g * CAPc + row0 + t] : -1;

    v8f zero = {0.f, 0.f, 0.f, 0.f, 0.f, 0.f, 0.f, 0.f};
    v8f acc[4][2];
    #pragma unroll
    for (int h = 0; h < 4; ++h) { acc[h][0] = zero; acc[h][1] = zero; }

    const unsigned short* __restrict__ Whg = Wh + (size_t)g * DIN * DOUT;

    for (int kc8 = 0; kc8 < 8; ++kc8) {
        const int kc = kc8 * 32;
        __syncthreads();   // covers eid init on first iter + LDS reuse after

        // ---- stage B chunk: contiguous 16 KB of pre-swizzled fp16 W
        {
            const uint4* src = (const uint4*)Whg + (size_t)kc8 * 1024 + t;
            uint4*       dst = (uint4*)&Bs[0][0][0] + t;
#if USE_ASYNC_LDS
            #pragma unroll
            for (int i = 0; i < 4; ++i)
                __builtin_amdgcn_global_load_async_to_lds_b128(
                    (gas_v4i)(src + 256 * i),
                    (las_v4i)(dst + 256 * i),
                    0, 0);
#else
            #pragma unroll
            for (int i = 0; i < 4; ++i) dst[256 * i] = src[256 * i];
#endif
        }

        // ---- stage A chunk: 64 rows x 32 k, gathered, fp32->fp16, packed b128 store
        {
            int r  = t >> 2;            // 0..63
            int k0 = (t & 3) * 8;       // 0,8,16,24
            float4 va = make_float4(0.f, 0.f, 0.f, 0.f);
            float4 vb = make_float4(0.f, 0.f, 0.f, 0.f);
            int e_ = eid[r];
            if (e_ >= 0) {
                const float* p = x + (size_t)e_ * DIN + kc + k0;
                va = *(const float4*)p;
                vb = *(const float4*)(p + 4);
            }
            // A-frag map (contiguous k = 16*lhk + slot, matches B)
            int h   = r >> 4, rl = r & 15;
            int lhk = k0 >> 4;                  // k0=0,8 -> 0 ; k0=16,24 -> 1
            int s0  = k0 & 15;                  // slot base 0 or 8
            uint4 u;
            u.x = pk2h(va.x, va.y); u.y = pk2h(va.z, va.w);
            u.z = pk2h(vb.x, vb.y); u.w = pk2h(vb.z, vb.w);
            *(uint4*)&As[h][lhk * 16 + rl][s0] = u;
        }

#if USE_ASYNC_LDS
        __builtin_amdgcn_s_wait_asynccnt(0);
#endif
        __syncthreads();

        // ---- fragments straight out of LDS (contiguous 32B per lane) + 8x WMMA
        v16h b0 = *(const v16h*)&Bs[2 * wv + 0][lane][0];
        v16h b1 = *(const v16h*)&Bs[2 * wv + 1][lane][0];
        #pragma unroll
        for (int h = 0; h < 4; ++h) {
            v16h a = *(const v16h*)&As[h][lane][0];
            acc[h][0] = __builtin_amdgcn_wmma_f32_16x16x32_f16(false, a, false, b0,
                                                               (short)0, acc[h][0], false, false);
            acc[h][1] = __builtin_amdgcn_wmma_f32_16x16x32_f16(false, a, false, b1,
                                                               (short)0, acc[h][1], false, false);
        }
    }

    // ---- epilogue: bias + scatter back to edge order
    const int col0 = wv * 32 + ln;
    const float bz0 = bias[g * DOUT + col0];
    const float bz1 = bias[g * DOUT + col0 + 16];
    #pragma unroll
    for (int h = 0; h < 4; ++h) {
        #pragma unroll
        for (int v = 0; v < 8; ++v) {
            int r = h * 16 + v + 8 * lh;        // C/D layout: lane half owns rows m/ m+8
            if (r < rows) {
                size_t base = (size_t)eid[r] * DOUT;
                out[base + col0]      = acc[h][0][v] + bz0;
                out[base + col0 + 16] = acc[h][1][v] + bz1;
            }
        }
    }
}

// ---------------- launcher ----------------
extern "C" void kernel_launch(void* const* d_in, const int* in_sizes, int n_in,
                              void* d_out, int out_size, void* d_ws, size_t ws_size,
                              hipStream_t stream) {
    const float* edge_feats = (const float*)d_in[0];
    const float* node_type  = (const float*)d_in[1];
    const float* edge_types = (const float*)d_in[2];
    const int*   edge_index = (const int*)  d_in[3];
    const float* W          = (const float*)d_in[4];
    const float* b          = (const float*)d_in[5];
    float*       out        = (float*)d_out;

    int* counts  = (int*)d_ws;                                     // 72 ints (pad 128)
    int* buckets = counts + 128;                                   // 72*2560 ints
    unsigned short* Wh = (unsigned short*)((char*)d_ws + (1u << 20)); // @1MB, 9.44MB fp16

    zero_counts_kernel<<<1, 128, 0, stream>>>(counts);
    route_kernel<<<(EE + 255) / 256, 256, 0, stream>>>(node_type, edge_types, edge_index,
                                                       counts, buckets);
    wconvert_kernel<<<(GG * (DIN / 2) * (DOUT / 4) + 255) / 256, 256, 0, stream>>>(W, Wh);
    dim3 grid(CAPc / TM, GG);             // 40 x 72
    grouped_gemm_kernel<<<grid, 256, 0, stream>>>(edge_feats, Wh, b, counts, buckets, out);

    (void)in_sizes; (void)n_in; (void)out_size; (void)ws_size;
}
